// Correlation_4166118277406
// MI455X (gfx1250) — matrix operand order repaired and checked by reading, not compiled, not run on hardware
//
#include <hip/hip_runtime.h>

// Cost-volume correlation, B=8 C=128 H=W=128, max_disp=4 (81 displacements).
//
// Phase 1a: zero-fill padded in2 workspace.
// Phase 1b: pack in1 NCHW f32 -> NHWC f16 [B,H,W,C]            (32 MB)
//           pack in2 NCHW f32 -> padded NHWC f16 [B,136,144,C] (~40 MB)
//           (both fit the 192 MB L2 together, so phase 2 streams from L2)
// Phase 2:  per (b, y, 16-wide x-tile) wave, branch-free all-pairs tiles
//           D[x1,x2] = sum_c in1[c,x1]*in2[c,x2] via v_wmma_f32_16x16x32_f16
//           (two 16x16 accumulators cover the 24-wide x2 window). All 8 B
//           operands are loaded before the 8 WMMAs so the scheduler can emit
//           one load clause and overlap waits with matrix ops. D is spilled
//           to LDS, the 9 dx diagonals extracted, scaled by 1/128, stored to
//           the 81 output planes. Physical zero padding reproduces the
//           reference's ConstantPad2d semantics with no branches.

typedef __attribute__((ext_vector_type(16))) _Float16 v16h;
typedef __attribute__((ext_vector_type(8)))  _Float16 v8h;
typedef __attribute__((ext_vector_type(8)))  float    v8f;

#define CB 8
#define CC 128
#define CH 128
#define CW 128
#define HP (CH + 8)    // padded rows: y2 in [-4, 131] -> yp = y2+4 in [0, 136)
#define WP 144         // padded cols: x2 in [-4, 139] -> xp = x2+4 in [0, 144)

__global__ void __launch_bounds__(256)
zero_fill_u32(unsigned int* __restrict__ dst) {
  dst[blockIdx.x * blockDim.x + threadIdx.x] = 0u;
}

__global__ void __launch_bounds__(256)
pack1_nhwc_f16(const float* __restrict__ src, _Float16* __restrict__ dst) {
  const int idx = blockIdx.x * blockDim.x + threadIdx.x;   // coalesced read over x
  const int x = idx & (CW - 1);
  const int y = (idx >> 7) & (CH - 1);
  const int c = (idx >> 14) & (CC - 1);
  const int b = idx >> 21;
  dst[((((size_t)b * CH + y) * CW + x) << 7) + c] = (_Float16)src[idx];
}

__global__ void __launch_bounds__(256)
pack2_padded_nhwc_f16(const float* __restrict__ src, _Float16* __restrict__ dst) {
  const int idx = blockIdx.x * blockDim.x + threadIdx.x;
  const int x = idx & (CW - 1);
  const int y = (idx >> 7) & (CH - 1);
  const int c = (idx >> 14) & (CC - 1);
  const int b = idx >> 21;
  dst[((((size_t)b * HP + (y + 4)) * WP + (x + 4)) << 7) + c] = (_Float16)src[idx];
}

__global__ void __launch_bounds__(256)
corr_wmma(const _Float16* __restrict__ in1h, const _Float16* __restrict__ in2p,
          float* __restrict__ out) {
  __shared__ float ldsD[8][512];                 // per-wave 16x32 f32 pair tile

  const int lane = threadIdx.x & 31;
  const int wave = threadIdx.x >> 5;
  const int tile = blockIdx.x * 8 + wave;        // 1024 blocks * 8 waves = 8192 tiles
  const int x0 = (tile & 7) << 4;                // x-tile origin
  const int y  = (tile >> 3) & 127;
  const int b  = tile >> 10;
  const int n    = lane & 15;                    // matrix row/col owned by this lane
  const int half = lane >> 4;

  // --- Preload A: in1 tile, 16 pixels x 128 channels, contiguous k order.
  // lane holds M = n; element i -> c = kk*32 + half*16 + i.
  v16h a[4];
  {
    const _Float16* aBase = in1h + ((((size_t)b * CH + y) * CW + (x0 + n)) << 7);
#pragma unroll
    for (int kk = 0; kk < 4; ++kk) {
      ((v8h*)&a[kk])[0] = *(const v8h*)(aBase + kk * 32 + half * 16);
      ((v8h*)&a[kk])[1] = *(const v8h*)(aBase + kk * 32 + half * 16 + 8);
    }
  }

  float* myD = ldsD[wave];
  const size_t plane = (size_t)CH * CW;
  const size_t outBase = (size_t)b * 81 * plane + (size_t)y * CW + x0;

  for (int dyi = 0; dyi < 9; ++dyi) {
    // Padded row yp = y + dyi covers y2 = y + dyi - 4; pad rows are zeros.
    const _Float16* row2 = in2p + ((((size_t)b * HP + (y + dyi)) * WP) << 7);
    // B-tile base for this lane: tile1 col n -> xp = x0+n; tile2 -> xp+16.
    const _Float16* p1base = row2 + (((size_t)(x0 + n)) << 7) + half * 16;

    // Warm WGP$ for the next dy row while this iteration computes.
    if (dyi < 8) {
      __builtin_prefetch(p1base + ((size_t)WP << 7), 0, 3);
      __builtin_prefetch(p1base + ((size_t)WP << 7) + (16 << 7), 0, 3);
    }

    // Load all 8 B operand tiles first (distinct registers -> one big clause,
    // waits overlap with the WMMA drain below).
    // B operand, same contiguous k order as A: lane holds N = n; element i -> c = kk*32 + half*16 + i.
    v16h bt[8];
#pragma unroll
    for (int kk = 0; kk < 4; ++kk) {
      const _Float16* p1 = p1base + kk * 32;
      const _Float16* p2 = p1 + (16 << 7);       // +16 padded columns
      ((v8h*)&bt[kk])[0]     = *(const v8h*)(p1);
      ((v8h*)&bt[kk])[1]     = *(const v8h*)(p1 + 8);
      ((v8h*)&bt[4 + kk])[0] = *(const v8h*)(p2);
      ((v8h*)&bt[4 + kk])[1] = *(const v8h*)(p2 + 8);
    }

    v8f d1 = {}; v8f d2 = {};
#pragma unroll
    for (int kk = 0; kk < 4; ++kk) {
      d1 = __builtin_amdgcn_wmma_f32_16x16x32_f16(false, a[kk], false, bt[kk],
                                                  (short)0, d1, false, false);
      d2 = __builtin_amdgcn_wmma_f32_16x16x32_f16(false, a[kk], false, bt[4 + kk],
                                                  (short)0, d2, false, false);
    }

    // D layout: VGPR r -> M = r + 8*half, N = n. Spill [16 x 32] pair tile to LDS.
#pragma unroll
    for (int r = 0; r < 8; ++r) {
      myD[(r + 8 * half) * 32 + n]      = d1[r];
      myD[(r + 8 * half) * 32 + 16 + n] = d2[r];
    }
    asm volatile("s_wait_dscnt 0" ::: "memory"); // same-wave DS write->read order

    if (lane < 16) {
      const int i = lane;                        // pixel x0+i, D row i
#pragma unroll
      for (int dxi = 0; dxi < 9; ++dxi) {        // column j = i + dxi (x2 = x+dxi-4)
        const float v = myD[i * 32 + i + dxi] * (1.0f / 128.0f);
        out[outBase + (size_t)(dyi * 9 + dxi) * plane + i] = v;
      }
    }
  }
}

extern "C" void kernel_launch(void* const* d_in, const int* in_sizes, int n_in,
                              void* d_out, int out_size, void* d_ws, size_t ws_size,
                              hipStream_t stream) {
  const float* in1 = (const float*)d_in[0];
  const float* in2 = (const float*)d_in[1];
  float* out = (float*)d_out;

  const int N  = CB * CC * CH * CW;              // 16,777,216 per input
  const int NP = CB * HP * WP * CC;              // 20,054,016 padded in2 elements
  _Float16* ws1  = (_Float16*)d_ws;              // in1 packed  : N  halfs (32 MB)
  _Float16* ws2p = ws1 + N;                      // in2 padded  : NP halfs (~40 MB)

  // Zero the padded in2 buffer (as u32 stores), then pack interiors over it.
  zero_fill_u32<<<(NP / 2) / 256, 256, 0, stream>>>((unsigned int*)ws2p);
  pack1_nhwc_f16<<<N / 256, 256, 0, stream>>>(in1, ws1);
  pack2_padded_nhwc_f16<<<N / 256, 256, 0, stream>>>(in2, ws2p);

  // 8192 wave-tiles: 8 waves/block * 1024 blocks, one wave per (b, y, x-tile).
  corr_wmma<<<1024, 256, 0, stream>>>(ws1, ws2p, out);
}
